// LLamaAttentionBlock_7834020348283
// MI455X (gfx1250) — matrix operand order repaired and checked by reading, not compiled, not run on hardware
//
#include <hip/hip_runtime.h>
#include <hip/hip_bf16.h>

// ---------------- problem constants (from reference) ----------------
constexpr int kNH   = 32;            // heads
constexpr int kDH   = 128;           // head dim
constexpr int kDM   = 4096;          // model dim
constexpr int kB    = 8;             // batch
constexpr int kSQ   = 16;            // new tokens
constexpr int kSKV  = 4096;          // cache length
constexpr int kSK   = kSKV + kSQ;    // total keys = 4112
constexpr int kROWS = kB * kSQ;      // 128 flattened query rows
constexpr float kSCALE = 0.08838834764831845f; // 1/sqrt(128)
constexpr int kNW = 8;               // waves per attention block

typedef __attribute__((ext_vector_type(16))) __bf16 bf16x16;
typedef __attribute__((ext_vector_type(8)))  float  f32x8;

__device__ __forceinline__ float rowmax16(float x) {
  x = fmaxf(x, __shfl_xor(x, 1, 32));
  x = fmaxf(x, __shfl_xor(x, 2, 32));
  x = fmaxf(x, __shfl_xor(x, 4, 32));
  x = fmaxf(x, __shfl_xor(x, 8, 32));
  return x;
}
__device__ __forceinline__ float rowsum16(float x) {
  x += __shfl_xor(x, 1, 32);
  x += __shfl_xor(x, 2, 32);
  x += __shfl_xor(x, 4, 32);
  x += __shfl_xor(x, 8, 32);
  return x;
}

// ================= Kernel 1: fused QKV projections ==================
// grid = (kDM/128, 1, 3), block = 256 (8 waves).
// Each wave owns a full-M column stripe: 8 stacked 16x16 output tiles
// (8 f32 accumulators = 64 VGPRs). Per 32-wide K step it loads ONE
// B fragment of W and reuses it across 8 v_wmma_f32_16x16x32_bf16 —
// weights stream from HBM exactly once (no M-slab re-reads).
__global__ __launch_bounds__(256) void proj_gemm_kernel(
    const float* __restrict__ Q, const float* __restrict__ K,
    const float* __restrict__ V, const float* __restrict__ w1,
    const float* __restrict__ w2, const float* __restrict__ w3,
    float* __restrict__ qkv)
{
  const int which = blockIdx.z;
  const float* X = (which == 0) ? Q : (which == 1) ? K : V;
  const float* W = (which == 0) ? w1 : (which == 1) ? w2 : w3;
  float* Out = qkv + (size_t)which * kROWS * kDM;

  const int wave  = threadIdx.x >> 5;
  const int lane  = threadIdx.x & 31;
  const int lrow  = lane & 15;   // A row / B col / C col within tile
  const int lhalf = lane >> 4;   // lane half selects K/M sub-ranges

  const int n0 = blockIdx.x * 128 + wave * 16;
  const float* wcol = W + n0 + lrow;                  // B: column of W
  const float* xrow = X + (size_t)lrow * kDM;         // A: row-major read

  const f32x8 zero8 = {0.f,0.f,0.f,0.f,0.f,0.f,0.f,0.f};
  f32x8 acc[8];
#pragma unroll
  for (int si = 0; si < 8; ++si) acc[si] = zero8;

  for (int kk = 0; kk < kDM; kk += 32) {
    // B fragment in this code's contiguous k order: VGPR v holds K = 16*half + 2v (+1), col = lane%16
    bf16x16 b;
#pragma unroll
    for (int v = 0; v < 8; ++v) {
      const int kb = kk + 16 * lhalf + 2 * v;
      b[2*v]   = (__bf16)wcol[(size_t)kb * kDM];
      b[2*v+1] = (__bf16)wcol[(size_t)(kb + 1) * kDM];
    }
    // 8 M-slabs reuse the same B fragment
#pragma unroll
    for (int si = 0; si < 8; ++si) {
      const float* xr = xrow + (size_t)(si * 16) * kDM;
      bf16x16 a;
#pragma unroll
      for (int v = 0; v < 8; ++v) {
        // A fragment in this code's contiguous k order: VGPR v holds K = 16*half + 2v (+1)
        const int g = v >> 2, j2 = v & 3;
        const int da = kk + 16 * lhalf + 2 * v;
        a[2*v]   = (__bf16)xr[da];
        a[2*v+1] = (__bf16)xr[da + 1];
      }
      acc[si] = __builtin_amdgcn_wmma_f32_16x16x32_bf16(false, a, false, b,
                                                        (short)0, acc[si], false, false);
    }
  }
  // C/D layout: VGPR v -> row 8*half+v, col = lane%16
#pragma unroll
  for (int si = 0; si < 8; ++si)
#pragma unroll
    for (int v = 0; v < 8; ++v)
      Out[(size_t)(si * 16 + 8 * lhalf + v) * kDM + n0 + lrow] = acc[si][v];
}

// ============ Kernel 2: flash attention over cache + new ============
// grid = B*H blocks, 256 threads (8 waves). Each wave handles a disjoint
// subset of 32-key blocks with online softmax; partials merged via LDS.
__global__ __launch_bounds__(256) void attn_kernel(
    const float* __restrict__ qproj, const float* __restrict__ knew,
    const float* __restrict__ vnew,  const float* __restrict__ Kcache,
    const float* __restrict__ Vcache, float* __restrict__ out)
{
  __shared__ float lds_O[kNW][16][kDH];   // per-wave O partials (merge stage)
  __shared__ float lds_m[kNW][16];
  __shared__ float lds_l[kNW][16];

  const int b = blockIdx.x / kNH;
  const int h = blockIdx.x % kNH;
  const int wave  = threadIdx.x >> 5;
  const int lane  = threadIdx.x & 31;
  const int lrow  = lane & 15;
  const int lhalf = lane >> 4;

  const float* qp = qproj  + ((size_t)b * kSQ)  * kDM + (size_t)h * kDH;
  const float* Kn = knew   + ((size_t)b * kSQ)  * kDM + (size_t)h * kDH;
  const float* Vn = vnew   + ((size_t)b * kSQ)  * kDM + (size_t)h * kDH;
  const float* Kc = Kcache + ((size_t)b * kSKV) * kDM + (size_t)h * kDH;
  const float* Vc = Vcache + ((size_t)b * kSKV) * kDM + (size_t)h * kDH;

  // wave-local P staging aliases this wave's O region (O only written after
  // the last P use; per-wave LDS ops are in-order on CDNA5)
  float (*ldsP)[32] = reinterpret_cast<float (*)[32]>(&lds_O[wave][0][0]);

  const f32x8 zero8 = {0.f,0.f,0.f,0.f,0.f,0.f,0.f,0.f};

  // ---- load Q as four 16x32 bf16 A-fragments (Dh = 128) ----
  bf16x16 qa[4];
  {
    const float* qrow = qp + (size_t)lrow * kDM;
#pragma unroll
    for (int s = 0; s < 4; ++s)
#pragma unroll
      for (int v = 0; v < 8; ++v) {
        const int g = v >> 2, j2 = v & 3;
        const int d = 32 * s + 16 * lhalf + 2 * v;
        qa[s][2*v]   = (__bf16)qrow[d];
        qa[s][2*v+1] = (__bf16)qrow[d + 1];
      }
  }

  f32x8 Oacc[8];
#pragma unroll
  for (int t = 0; t < 8; ++t) Oacc[t] = zero8;
  float mrow[8], lsum[8];
#pragma unroll
  for (int v = 0; v < 8; ++v) { mrow[v] = -1e30f; lsum[v] = 0.f; }

  const int nblk = (kSK + 31) / 32;   // 129: 128 cache blocks + partial new
  for (int bid = wave; bid < nblk; bid += kNW) {
    const int kb0 = bid * 32;

    // ---- scores S = q . k  (two 16-key tiles, contraction over Dh=128) ----
    f32x8 s[2];
#pragma unroll
    for (int half2 = 0; half2 < 2; ++half2) {
      const int j = kb0 + 16 * half2 + lrow;     // this lane's key column
      const bool valid = j < kSK;
      const float* kr = (j < kSKV) ? (Kc + (size_t)j * kDM)
                                   : (Kn + (size_t)(j - kSKV) * kDM);
      f32x8 acc = zero8;
#pragma unroll
      for (int sl = 0; sl < 4; ++sl) {
        bf16x16 kf;  // B fragment: col = key, K(=d) = 32*sl + 16*half + 2v
#pragma unroll
        for (int v = 0; v < 8; ++v) {
          const int d = 32 * sl + 16 * lhalf + 2 * v;
          kf[2*v]   = (__bf16)(valid ? kr[d]     : 0.f);
          kf[2*v+1] = (__bf16)(valid ? kr[d + 1] : 0.f);
        }
        acc = __builtin_amdgcn_wmma_f32_16x16x32_bf16(false, qa[sl], false, kf,
                                                      (short)0, acc, false, false);
      }
      s[half2] = acc;
    }

    // ---- scale + causal/bounds mask + online softmax update ----
    float alpha[8];
#pragma unroll
    for (int v = 0; v < 8; ++v) {
      const int m  = 8 * lhalf + v;        // query row of this element
      const int j0 = kb0 + lrow;
      const int j1 = kb0 + 16 + lrow;
      // key j visible iff j <= kSKV + m (covers both OOB and causal mask)
      const float s0 = (j0 <= kSKV + m) ? s[0][v] * kSCALE : -1e30f;
      const float s1 = (j1 <= kSKV + m) ? s[1][v] * kSCALE : -1e30f;
      const float bmax = rowmax16(fmaxf(s0, s1));
      const float mnew = fmaxf(mrow[v], bmax);
      alpha[v] = __expf(mrow[v] - mnew);
      const float p0 = __expf(s0 - mnew);
      const float p1 = __expf(s1 - mnew);
      lsum[v] = lsum[v] * alpha[v] + rowsum16(p0) + rowsum16(p1);
      mrow[v] = mnew;
      // stash P (C-fragment layout) for re-layout into an A-fragment
      ldsP[8 * lhalf + v][lrow]      = p0;
      ldsP[8 * lhalf + v][16 + lrow] = p1;
    }
#pragma unroll
    for (int t = 0; t < 8; ++t)
#pragma unroll
      for (int v = 0; v < 8; ++v)
        Oacc[t][v] *= alpha[v];

    __asm__ volatile("" ::: "memory");  // keep LDS store->load order

    // ---- read P back as a 16x32 bf16 A-fragment ----
    bf16x16 pa;
#pragma unroll
    for (int v = 0; v < 8; ++v) {
      const int g = v >> 2, j2 = v & 3;
      const int kk = 16 * lhalf + 2 * v;
      pa[2*v]   = (__bf16)ldsP[lrow][kk];
      pa[2*v+1] = (__bf16)ldsP[lrow][kk + 1];
    }

    // ---- O += P @ V  (8 d-tiles of 16 columns) ----
    const float* vrow[16];
    bool vok[16];
#pragma unroll
    for (int k2 = 0; k2 < 16; ++k2) {
      const int j = kb0 + 16 * lhalf + k2;
      vok[k2]  = j < kSK;
      vrow[k2] = (j < kSKV) ? (Vc + (size_t)j * kDM)
                            : (Vn + (size_t)(j - kSKV) * kDM);
    }
#pragma unroll
    for (int t = 0; t < 8; ++t) {
      bf16x16 vf;  // B fragment: col = d, K(=key-in-block) = 16*half + 2v
      const int dcol = 16 * t + lrow;
#pragma unroll
      for (int v = 0; v < 8; ++v) {
        vf[2*v]   = (__bf16)(vok[2*v]     ? vrow[2*v][dcol]     : 0.f);
        vf[2*v+1] = (__bf16)(vok[2*v + 1] ? vrow[2*v + 1][dcol] : 0.f);
      }
      Oacc[t] = __builtin_amdgcn_wmma_f32_16x16x32_bf16(false, pa, false, vf,
                                                        (short)0, Oacc[t], false, false);
    }
  }

  // ---- publish per-wave partials (after last P staging use) ----
  __asm__ volatile("" ::: "memory");
#pragma unroll
  for (int t = 0; t < 8; ++t)
#pragma unroll
    for (int v = 0; v < 8; ++v)
      lds_O[wave][8 * lhalf + v][16 * t + lrow] = Oacc[t][v];
  if (lrow == 0) {
#pragma unroll
    for (int v = 0; v < 8; ++v) {
      lds_m[wave][8 * lhalf + v] = mrow[v];
      lds_l[wave][8 * lhalf + v] = lsum[v];
    }
  }
  __syncthreads();

  // ---- merge 8 wave partials, normalize, write output (deterministic) ----
  for (int e = threadIdx.x; e < 16 * kDH; e += blockDim.x) {
    const int row = e >> 7;
    const int d   = e & (kDH - 1);
    float M = -1e30f;
#pragma unroll
    for (int w = 0; w < kNW; ++w) M = fmaxf(M, lds_m[w][row]);
    float num = 0.f, den = 0.f;
#pragma unroll
    for (int w = 0; w < kNW; ++w) {
      const float f = __expf(lds_m[w][row] - M);
      num += f * lds_O[w][row][d];
      den += f * lds_l[w][row];
    }
    out[((size_t)b * kSQ + row) * kDM + (size_t)h * kDH + d] = num / den;
  }
}

// =========================== launcher ===============================
extern "C" void kernel_launch(void* const* d_in, const int* in_sizes, int n_in,
                              void* d_out, int out_size, void* d_ws, size_t ws_size,
                              hipStream_t stream) {
  (void)in_sizes; (void)n_in; (void)out_size; (void)ws_size;
  const float* Q  = (const float*)d_in[0];
  const float* K  = (const float*)d_in[1];
  const float* V  = (const float*)d_in[2];
  const float* Kc = (const float*)d_in[3];
  const float* Vc = (const float*)d_in[4];
  const float* w1 = (const float*)d_in[5];
  const float* w2 = (const float*)d_in[6];
  const float* w3 = (const float*)d_in[7];

  float* qkv = (float*)d_ws;                        // 3 * 128 * 4096 f32 = 6 MB
  float* qp  = qkv;
  float* kn  = qkv + (size_t)kROWS * kDM;
  float* vn  = qkv + 2 * (size_t)kROWS * kDM;

  proj_gemm_kernel<<<dim3(kDM / 128, 1, 3), 256, 0, stream>>>(
      Q, K, V, w1, w2, w3, qkv);
  attn_kernel<<<kB * kNH, 256, 0, stream>>>(qp, kn, vn, Kc, Vc, (float*)d_out);
}
